// VectorQuantizer_39548058862063
// MI455X (gfx1250) — matrix operand order repaired and checked by reading, not compiled, not run on hardware
//
#include <hip/hip_runtime.h>
#include <hip/hip_bf16.h>

#define NUM_CODES 8192
#define CODE_DIM  256
#define N_ROWS    16384            // B*L = 8*2048
#define KT_TOTAL  (NUM_CODES / 16) // 512 code tiles
#define BLOCK_ROWS 64
#define NBLOCKS   (N_ROWS / BLOCK_ROWS) // 256
#define ITERS     (KT_TOTAL / 2)   // 256 iterations, two spatial tiles each
#define LDS_STRIDE 264             // padded row stride (bf16 elems) -> conflict-free b128 reads

typedef __attribute__((ext_vector_type(16))) __bf16 v16bf;
typedef __attribute__((ext_vector_type(8)))  float  v8f;

union Frag16 {
  v16bf v;
  unsigned short u[16];
  uint4 q[2];
};

__device__ __forceinline__ unsigned short f2bf_rne(float f) {
  unsigned int u = __float_as_uint(f);
  u += 0x7FFFu + ((u >> 16) & 1u);           // round-to-nearest-even
  return (unsigned short)(u >> 16);
}
__device__ __forceinline__ float bf2f(unsigned short h) {
  return __uint_as_float(((unsigned int)h) << 16);
}

// Convert 32 f32 (8 float4) -> 32 bf16 hi + 32 bf16 lo, store as 4+4 b128.
__device__ __forceinline__ void stage_store(const float4* pf,
                                            unsigned short* dh,
                                            unsigned short* dl) {
#pragma unroll
  for (int j = 0; j < 4; ++j) {
    float4 a = pf[2 * j], b = pf[2 * j + 1];
    float v[8] = {a.x, a.y, a.z, a.w, b.x, b.y, b.z, b.w};
    unsigned int hp[4], lp[4];
#pragma unroll
    for (int q = 0; q < 4; ++q) {
      unsigned short h0 = f2bf_rne(v[2 * q]);
      unsigned short h1 = f2bf_rne(v[2 * q + 1]);
      unsigned short l0 = f2bf_rne(v[2 * q] - bf2f(h0));
      unsigned short l1 = f2bf_rne(v[2 * q + 1] - bf2f(h1));
      hp[q] = (unsigned int)h0 | ((unsigned int)h1 << 16);
      lp[q] = (unsigned int)l0 | ((unsigned int)l1 << 16);
    }
    *(uint4*)(dh + j * 8) = make_uint4(hp[0], hp[1], hp[2], hp[3]);
    *(uint4*)(dl + j * 8) = make_uint4(lp[0], lp[1], lp[2], lp[3]);
  }
}

// ---------------------------------------------------------------------------
// Kernel 1: exact f32 ||e||^2 per code
// ---------------------------------------------------------------------------
__global__ void __launch_bounds__(256)
vq_e2_kernel(const float* __restrict__ emb, float* __restrict__ e2) {
  __shared__ float red[256];
  const int k = blockIdx.x, t = threadIdx.x;
  float v = emb[(size_t)k * CODE_DIM + t];
  red[t] = v * v;
  __syncthreads();
  for (int s = 128; s > 0; s >>= 1) {
    if (t < s) red[t] += red[t + s];
    __syncthreads();
  }
  if (t == 0) e2[k] = red[0];
}

// ---------------------------------------------------------------------------
// Kernel 2: main — split-precision bf16 WMMA GEMM, software-pipelined B tiles
// Block: 256 threads (8 waves). Waves 0-3: row tiles 0-3, even code tiles;
// waves 4-7: same row tiles, odd code tiles. LDS B double-buffered in time.
// ---------------------------------------------------------------------------
__global__ void __launch_bounds__(256)
vq_main_kernel(const float* __restrict__ z_e, const float* __restrict__ emb,
               const float* __restrict__ e2g, float* __restrict__ out_zq,
               float* __restrict__ out_codes, float* __restrict__ partials) {
  // [temporal set][spatial tile][code][dim]
  __shared__ __align__(16) unsigned short ldsBhi[2][2][16][LDS_STRIDE];
  __shared__ __align__(16) unsigned short ldsBlo[2][2][16][LDS_STRIDE];

  const int t    = threadIdx.x;
  const int wave = t >> 5;
  const int lane = t & 31;
  const int half = lane >> 4;   // 16-lane half of the wave
  const int col  = lane & 15;
  const int rt   = wave & 3;    // row tile owned by this wave
  const int bi   = wave >> 2;   // spatial code-tile buffer this wave consumes
  const int row_base = blockIdx.x * BLOCK_ROWS + rt * 16;

  // ---- Load A tile (16 rows x 256 dims) once; split f32 -> bf16 hi/lo ----
  // A 16-bit 16x32 layout: lane row = lane%16; slots 0..7 <-> K = 8*half+s,
  // slots 8..15 <-> K = 16 + 8*half + (s-8).
  Frag16 ahi[8], alo[8];
  {
    const float* arow = z_e + (size_t)(row_base + col) * CODE_DIM;
#pragma unroll
    for (int ks = 0; ks < 8; ++ks) {
      const float* p0 = arow + ks * 32 + 8 * half; // slots 0..7
      const float* p1 = p0 + 16;                   // slots 8..15
      float4 f0 = *(const float4*)(p0);
      float4 f1 = *(const float4*)(p0 + 4);
      float4 f2 = *(const float4*)(p1);
      float4 f3 = *(const float4*)(p1 + 4);
      float vals[16] = {f0.x, f0.y, f0.z, f0.w, f1.x, f1.y, f1.z, f1.w,
                        f2.x, f2.y, f2.z, f2.w, f3.x, f3.y, f3.z, f3.w};
#pragma unroll
      for (int s = 0; s < 16; ++s) {
        unsigned short h = f2bf_rne(vals[s]);
        ahi[ks].u[s] = h;
        alo[ks].u[s] = f2bf_rne(vals[s] - bf2f(h));
      }
    }
  }

  float best[8];
  int   bidx[8];
#pragma unroll
  for (int v = 0; v < 8; ++v) { best[v] = 3.4e38f; bidx[v] = 0; }

  // staging mapping: threads 0..127 fill spatial tile 0, 128..255 tile 1
  const int sbuf = t >> 7;
  const int su   = t & 127;
  const int scl  = su >> 3;        // code_local 0..15
  const int sds  = (su & 7) * 32;  // dim start (32 dims per thread)

  // ---- Prologue: stage iteration 0 into set 0 ----------------------------
  {
    float4 pf[8];
    const float* src = emb + (size_t)(sbuf * 16 + scl) * CODE_DIM + sds;
#pragma unroll
    for (int j = 0; j < 8; ++j) pf[j] = *(const float4*)(src + j * 4);
    stage_store(pf, &ldsBhi[0][sbuf][scl][sds], &ldsBlo[0][sbuf][scl][sds]);
  }
  __syncthreads();

  for (int it = 0; it < ITERS; ++it) {
    const int set = it & 1;

    // -- issue next tile's global loads first: latency hides under WMMAs --
    float4 pf[8];
    const bool havePf = (it + 1 < ITERS);
    if (havePf) {
      const int ctS = 2 * (it + 1) + sbuf;
      const float* src = emb + (size_t)(ctS * 16 + scl) * CODE_DIM + sds;
#pragma unroll
      for (int j = 0; j < 8; ++j) pf[j] = *(const float4*)(src + j * 4);
    }
    const int code = (2 * it + bi) * 16 + col;
    const float e2v = e2g[code];

    // -- 16x16 dot tile: 8 K-steps x 3 WMMAs, two independent chains ------
    v8f c0 = {}, c1 = {};
#pragma unroll
    for (int ks = 0; ks < 8; ++ks) {
      // B 16-bit 32x16 layout: col = lane%16, slots 0..7 <-> K = 8*half + s, slots 8..15 <-> K = 16 + 8*half + (s-8)
      Frag16 bh, bl;
      const unsigned short* pb = &ldsBhi[set][bi][col][ks * 32 + 8 * half];
      const unsigned short* pl = &ldsBlo[set][bi][col][ks * 32 + 8 * half];
      bh.q[0] = *(const uint4*)(pb);
      bh.q[1] = *(const uint4*)(pb + 16);
      bl.q[0] = *(const uint4*)(pl);
      bl.q[1] = *(const uint4*)(pl + 16);
      if (ks & 1) {
        c1 = __builtin_amdgcn_wmma_f32_16x16x32_bf16(false, ahi[ks].v, false, bh.v, (short)0, c1, false, false);
        c1 = __builtin_amdgcn_wmma_f32_16x16x32_bf16(false, ahi[ks].v, false, bl.v, (short)0, c1, false, false);
        c1 = __builtin_amdgcn_wmma_f32_16x16x32_bf16(false, alo[ks].v, false, bh.v, (short)0, c1, false, false);
      } else {
        c0 = __builtin_amdgcn_wmma_f32_16x16x32_bf16(false, ahi[ks].v, false, bh.v, (short)0, c0, false, false);
        c0 = __builtin_amdgcn_wmma_f32_16x16x32_bf16(false, ahi[ks].v, false, bl.v, (short)0, c0, false, false);
        c0 = __builtin_amdgcn_wmma_f32_16x16x32_bf16(false, alo[ks].v, false, bh.v, (short)0, c0, false, false);
      }
    }
    v8f c = c0 + c1;

    // C layout: VGPR v holds row (v + 8*half), col = lane%16
#pragma unroll
    for (int v = 0; v < 8; ++v) {
      float s = e2v - 2.0f * c[v];     // dist minus constant ||x||^2
      if (s < best[v]) { best[v] = s; bidx[v] = code; }
    }

    // -- convert + store next tile into the other set ----------------------
    if (havePf) {
      stage_store(pf, &ldsBhi[set ^ 1][sbuf][scl][sds],
                      &ldsBlo[set ^ 1][sbuf][scl][sds]);
    }
    __syncthreads();
  }

  // ---- reuse B LDS space for the reduction scratch (barrier-separated) ---
  float* sScore = (float*)&ldsBhi[0][0][0][0];           // 8*16*16 floats
  int*   sIdx   = (int*)(sScore + 8 * 16 * 16);          // 8*16*16 ints
  int*   sWin   = (int*)(sIdx + 8 * 16 * 16);            // 64 ints
  float* sRed   = (float*)(sWin + BLOCK_ROWS);           // 256 floats

#pragma unroll
  for (int v = 0; v < 8; ++v) {
    sScore[(wave * 16 + v + 8 * half) * 16 + col] = best[v];
    sIdx[(wave * 16 + v + 8 * half) * 16 + col]   = bidx[v];
  }
  __syncthreads();

  // ---- per-row argmin across 16 lane-columns and the two code-half waves --
  if (t < BLOCK_ROWS) {
    const int rtt = t >> 4, rl = t & 15;
    float bs = 3.4e38f;
    int   bwin = 0;
#pragma unroll
    for (int w2 = 0; w2 < 2; ++w2) {
      const int wv = rtt + 4 * w2;
      for (int cc = 0; cc < 16; ++cc) {
        float s = sScore[(wv * 16 + rl) * 16 + cc];
        int   i = sIdx[(wv * 16 + rl) * 16 + cc];
        if (s < bs || (s == bs && i < bwin)) { bs = s; bwin = i; }
      }
    }
    sWin[t] = bwin;
    out_codes[blockIdx.x * BLOCK_ROWS + t] = (float)bwin;
  }
  __syncthreads();

  // ---- z_q gather (exact f32 from emb) + SSD partial ----------------------
  float ssd = 0.0f;
  {
    const int rl   = t >> 2;
    const int db   = (t & 3) * 64;
    const int grow = blockIdx.x * BLOCK_ROWS + rl;
    const int code = sWin[rl];
    const float4* ep = (const float4*)(emb + (size_t)code * CODE_DIM + db);
    const float4* zp = (const float4*)(z_e + (size_t)grow * CODE_DIM + db);
    float4* op = (float4*)(out_zq + (size_t)grow * CODE_DIM + db);
#pragma unroll 4
    for (int i = 0; i < 16; ++i) {
      float4 e = ep[i], z = zp[i];
      op[i] = e;   // z_q_st == z_q numerically (straight-through)
      float dx = e.x - z.x, dy = e.y - z.y, dz = e.z - z.z, dw = e.w - z.w;
      ssd += dx * dx + dy * dy + dz * dz + dw * dw;
    }
  }
  sRed[t] = ssd;
  __syncthreads();
  for (int s = 128; s > 0; s >>= 1) {
    if (t < s) sRed[t] += sRed[t + s];
    __syncthreads();
  }
  if (t == 0) partials[blockIdx.x] = sRed[0];
}

// ---------------------------------------------------------------------------
// Kernel 3: deterministic fixed-order loss finalize
// loss = embedding_loss + 0.25*commitment = 1.25 * mean((z_q - z_e)^2)
// ---------------------------------------------------------------------------
__global__ void vq_finalize_kernel(const float* __restrict__ partials,
                                   float* __restrict__ out_loss) {
  if (threadIdx.x == 0 && blockIdx.x == 0) {
    float s = 0.0f;
    for (int i = 0; i < NBLOCKS; ++i) s += partials[i];
    *out_loss = 1.25f * s / (float)((size_t)N_ROWS * CODE_DIM);
  }
}

extern "C" void kernel_launch(void* const* d_in, const int* in_sizes, int n_in,
                              void* d_out, int out_size, void* d_ws, size_t ws_size,
                              hipStream_t stream) {
  (void)in_sizes; (void)n_in; (void)out_size; (void)ws_size;
  const float* z_e = (const float*)d_in[0];
  const float* emb = (const float*)d_in[1];

  float* out       = (float*)d_out;
  float* out_zq    = out;                                   // (8,2048,256)
  float* out_loss  = out + (size_t)N_ROWS * CODE_DIM;       // scalar
  float* out_codes = out_loss + 1;                          // (8,2048)

  float* e2       = (float*)d_ws;          // 8192 floats
  float* partials = e2 + NUM_CODES;        // 256 floats

  vq_e2_kernel<<<NUM_CODES, 256, 0, stream>>>(emb, e2);
  vq_main_kernel<<<NBLOCKS, 256, 0, stream>>>(z_e, emb, e2, out_zq, out_codes,
                                              partials);
  vq_finalize_kernel<<<1, 64, 0, stream>>>(partials, out_loss);
}
